// SymmetricConfidenceBiasedMHA_3659312136406
// MI455X (gfx1250) — matrix operand order repaired and checked by reading, not compiled, not run on hardware
//
#include <hip/hip_runtime.h>
#include <hip/hip_bf16.h>

typedef __bf16 bf16;
typedef __attribute__((ext_vector_type(16))) __bf16 v16b;
typedef __attribute__((ext_vector_type(8)))  __bf16 v8b;
typedef __attribute__((ext_vector_type(8)))  float  v8f;
typedef unsigned int u32x4 __attribute__((ext_vector_type(4)));
typedef int          i32x8 __attribute__((ext_vector_type(8)));
typedef int          i32x4 __attribute__((ext_vector_type(4)));

#define BATCH  2
#define TSEQ   2048
#define DMODEL 1024
#define NHEAD  16
#define HDK    64
#define NTOK   (BATCH * TSEQ)   // 4096

// ---------------------------------------------------------------------------
// WMMA helpers (CDNA5 gfx1250, wave32)
// ---------------------------------------------------------------------------
__device__ __forceinline__ v8f wmma_bf16(v16b a, v16b b, v8f c) {
  // D = A(16x32 bf16) * B(32x16 bf16) + C(16x16 f32)
  return __builtin_amdgcn_wmma_f32_16x16x32_bf16(
      /*neg_a=*/false, a, /*neg_b=*/false, b,
      /*c_mod=*/(short)0, c, /*reuse_a=*/false, /*reuse_b=*/false);
}

// A fragment 16x32 (MxK): lane L holds row m=L&15; K-halves split by L>>4.
__device__ __forceinline__ v16b load_frag_a(const bf16* __restrict__ base,
                                            int ld, int lane) {
  const int m = lane & 15, hh = lane >> 4;
  return *(const v16b*)(base + (size_t)m * ld + hh * 16);
}

// B fragment 32x16 (KxN), where B[k][n] = src[n][k] (src row-major by n).
__device__ __forceinline__ v16b load_frag_b(const bf16* __restrict__ base,
                                            int ld, int lane) {
  const int n = lane & 15, hh = lane >> 4;
  return *(const v16b*)(base + (size_t)n * ld + hh * 16);
}

// ---------------------------------------------------------------------------
// Kernel 0: f32 -> bf16 conversion / packing (all operands become L2-resident)
// ---------------------------------------------------------------------------
__global__ void __launch_bounds__(256)
cvt_kernel(const float* __restrict__ x,
           const float* __restrict__ Wq, const float* __restrict__ Wk,
           const float* __restrict__ Wv, const float* __restrict__ Wo,
           bf16* __restrict__ xbf, bf16* __restrict__ wqkv,
           bf16* __restrict__ wobf) {
  const int i = blockIdx.x * blockDim.x + threadIdx.x;
  const int NX = NTOK * DMODEL;          // 4M
  const int NW = DMODEL * DMODEL;        // 1M
  if (i < NX) xbf[i] = (bf16)x[i];
  if (i < NW) {
    wqkv[i]          = (bf16)Wq[i];
    wqkv[NW + i]     = (bf16)Wk[i];
    wqkv[2 * NW + i] = (bf16)Wv[i];
    wobf[i]          = (bf16)Wo[i];
  }
}

// ---------------------------------------------------------------------------
// Kernel 1: fused QKV projection GEMM  [4096 x 1024] x [1024 x 3072]
// Block: 256 thr = 8 waves arranged 2(M) x 4(N); wave tile 32x64.
// Q,K stored [B,H,T,DK]; V stored transposed [B,H,DK,T] (+bias) for PV GEMM.
// ---------------------------------------------------------------------------
__global__ void __launch_bounds__(256)
proj_qkv(const bf16* __restrict__ xbf, const bf16* __restrict__ wqkv,
         const float* __restrict__ bv,
         bf16* __restrict__ Qd, bf16* __restrict__ Kd, bf16* __restrict__ Vtd) {
  const int lane = threadIdx.x & 31, wave = threadIdx.x >> 5;
  const int mw = wave >> 2;                  // 0..1
  const int nw = wave & 3;                   // 0..3
  const int m0 = blockIdx.x * 64 + mw * 32;  // token base (2 tiles of 16)
  const int n0 = blockIdx.y * 256 + nw * 64; // feature base (4 tiles of 16)

  v8f acc[2][4] = {};
  for (int k = 0; k < DMODEL; k += 32) {
    if (k + 32 < DMODEL) {  // warm WGP$ for next K-step (global_prefetch_b8)
      __builtin_prefetch((const void*)(xbf + (size_t)m0 * DMODEL + k + 32), 0, 3);
      __builtin_prefetch((const void*)(wqkv + (size_t)n0 * DMODEL + k + 32), 0, 3);
    }
    v16b a0 = load_frag_a(xbf + (size_t)m0 * DMODEL + k, DMODEL, lane);
    v16b a1 = load_frag_a(xbf + (size_t)(m0 + 16) * DMODEL + k, DMODEL, lane);
#pragma unroll
    for (int j = 0; j < 4; ++j) {
      v16b b = load_frag_b(wqkv + (size_t)(n0 + j * 16) * DMODEL + k,
                           DMODEL, lane);
      acc[0][j] = wmma_bf16(a0, b, acc[0][j]);
      acc[1][j] = wmma_bf16(a1, b, acc[1][j]);
    }
  }

  const int hh = lane >> 4, nn = lane & 15;
#pragma unroll
  for (int mi = 0; mi < 2; ++mi) {
#pragma unroll
    for (int j = 0; j < 4; ++j) {
      const int n = n0 + j * 16 + nn;
      const int proj = n >> 10;           // 0=Q 1=K 2=V
      const int f = n & 1023;
      const int h = f >> 6, dk = f & 63;
#pragma unroll
      for (int v = 0; v < 8; ++v) {
        const int m = m0 + mi * 16 + hh * 8 + v;
        const int b = m >> 11, t = m & 2047;
        const float val = acc[mi][j][v];
        const size_t bh = (size_t)b * NHEAD + h;
        if (proj == 0) {
          Qd[(bh * TSEQ + t) * HDK + dk] = (bf16)val;
        } else if (proj == 1) {
          Kd[(bh * TSEQ + t) * HDK + dk] = (bf16)val;
        } else {
          Vtd[(bh * HDK + dk) * TSEQ + t] = (bf16)(val + bv[f]);
        }
      }
    }
  }
}

// ---------------------------------------------------------------------------
// Kernel 2: attention.  One block per (b, h, 16-query tile).
// LDS: 2KB Q tile (TDM-staged) + 16x2048 bf16 scores/P (64KB) + reduce bufs.
// ~72KB/block -> ~4 workgroups per 320KB WGP (vs 1 before) for latency hiding.
// Exact softmax (full bf16 score row in LDS).
// ---------------------------------------------------------------------------
#define SQ_BYTES   (16 * HDK * 2)                       // 2048
#define SP_BYTES   (16 * TSEQ * 2)                      // 65536
#define SOUT_BYTES (16 * 64 * 4)                        // 4096
#define ATTN_SMEM  (SQ_BYTES + SP_BYTES + SOUT_BYTES + 64 + 64)

__global__ void __launch_bounds__(256)
attn_kernel(const bf16* __restrict__ Qd, const bf16* __restrict__ Kd,
            const bf16* __restrict__ Vtd,
            const float* __restrict__ conf, const float* __restrict__ Wc,
            const float* __restrict__ bc, const float* __restrict__ alphap,
            bf16* __restrict__ attnb) {
  extern __shared__ char smem[];
  bf16*  sQ   = (bf16*)(smem);                          // 16x64 bf16 @ offset 0
  bf16*  sP   = (bf16*)(smem + SQ_BYTES);               // 16x2048 bf16
  float* sOut = (float*)(smem + SQ_BYTES + SP_BYTES);   // 16x64 f32
  float* sRow = sOut + 16 * 64;                         // 16 f32
  float* sCbQ = sRow + 16;                              // 16 f32

  const int bid = blockIdx.x;
  const int qt = bid & 127;            // T/16 = 128 query tiles
  const int bh = bid >> 7;             // 0..31
  const int h = bh & 15, b = bh >> 4;
  const int t0 = qt * 16;

  const int lane = threadIdx.x & 31, wave = threadIdx.x >> 5;
  const int hh = lane >> 4, nn = lane & 15;

  const float a_sig = 1.0f / (1.0f + __expf(-alphap[0]));
  const float wc = Wc[h], bch = bc[h];
  const float inv_scale = 0.125f;      // 1/sqrt(64)

  const bf16* Qb = Qd + ((size_t)bh * TSEQ + t0) * HDK;
  const bf16* Kb = Kd + (size_t)bh * TSEQ * HDK;
  const bf16* Vb = Vtd + (size_t)bh * HDK * TSEQ;

  // --- stage Q tile (16x64 bf16, contiguous 2KB) into LDS via the TDM ---
#if defined(__has_builtin) && __has_builtin(__builtin_amdgcn_tensor_load_to_lds)
  if (wave == 0) {
    const unsigned long long ga = (unsigned long long)(uintptr_t)Qb;
    u32x4 g0;
    g0.x = 1u;                                  // count=1, user (not restore)
    g0.y = 0u;                                  // lds_addr = 0 (sQ at base)
    g0.z = (unsigned)(ga & 0xFFFFFFFFu);        // global_addr[31:0]
    g0.w = (unsigned)((ga >> 32) & 0x01FFFFFFu) // global_addr[56:32]
         | (2u << 30);                          // D#.type = 2 ("image")
    i32x8 g1;
    g1[0] = 0x00010000;                         // data_size=1 (2B), mask=0
    g1[1] = (int)(64u << 16);                   // tensor_dim0 = 64 (lo16)
    g1[2] = (int)(2048u << 16);                 // dim0 hi16=0 | tensor_dim1 lo16
    g1[3] = (int)(64u << 16);                   // dim1 hi16=0 | tile_dim0=64
    g1[4] = 16;                                 // tile_dim1=16, tile_dim2=0
    g1[5] = 64;                                 // tensor_dim0_stride = 64
    g1[6] = 0;                                  // dim0_stride hi | dim1_stride lo
    g1[7] = 0;
    i32x4 z4 = {0, 0, 0, 0};                    // groups 2/3 unused (<=2-D)
#if __clang_major__ >= 23
    i32x8 z8 = {};
    __builtin_amdgcn_tensor_load_to_lds(g0, g1, z4, z4, z8, 0);
#else
    __builtin_amdgcn_tensor_load_to_lds(g0, g1, z4, z4, 0);
#endif
    __builtin_amdgcn_s_wait_tensorcnt(0);
  }
#else
  for (int i = threadIdx.x; i < 16 * HDK; i += 256) sQ[i] = Qb[i];
#endif

  if (threadIdx.x < 16)
    sCbQ[threadIdx.x] =
        a_sig * (conf[b * TSEQ + t0 + threadIdx.x] * wc + bch);
  for (int i = threadIdx.x; i < 16 * 64; i += 256) sOut[i] = 0.0f;
  __syncthreads();

  // Q tile fragments from LDS (reused by every wave for all 128 key tiles)
  v16b aq0 = load_frag_a(sQ + 0, HDK, lane);
  v16b aq1 = load_frag_a(sQ + 32, HDK, lane);

  // --- QK^T: each wave builds 16 key tiles of the 16x2048 score strip ---
  for (int jt = wave * 16; jt < wave * 16 + 16; ++jt) {
    const int kb = jt * 16;
    v16b bk0 = load_frag_b(Kb + (size_t)kb * HDK + 0, HDK, lane);
    v16b bk1 = load_frag_b(Kb + (size_t)kb * HDK + 32, HDK, lane);
    v8f c = {};
    c = wmma_bf16(aq0, bk0, c);
    c = wmma_bf16(aq1, bk1, c);
    const float cbk = a_sig * (conf[b * TSEQ + kb + nn] * wc + bch);
#pragma unroll
    for (int v = 0; v < 8; ++v) {
      const int m = hh * 8 + v;
      sP[m * TSEQ + kb + nn] = (bf16)(c[v] * inv_scale + sCbQ[m] + cbk);
    }
  }
  __syncthreads();

  // --- exact row softmax over bf16 scores in LDS; 16 lanes per row ---
  {
    const int r = threadIdx.x >> 4;    // 0..15
    const int sub = threadIdx.x & 15;
    float mx = -1e30f;
    for (int j = sub; j < TSEQ; j += 16)
      mx = fmaxf(mx, (float)sP[r * TSEQ + j]);
#pragma unroll
    for (int o = 8; o >= 1; o >>= 1) mx = fmaxf(mx, __shfl_xor(mx, o, 16));
    float sum = 0.0f;
    for (int j = sub; j < TSEQ; j += 16) {
      const float e = __expf((float)sP[r * TSEQ + j] - mx);
      sum += e;
      sP[r * TSEQ + j] = (bf16)e;      // unnormalized; fold 1/sum at epilogue
    }
#pragma unroll
    for (int o = 8; o >= 1; o >>= 1) sum += __shfl_xor(sum, o, 16);
    if (sub == 0) sRow[r] = 1.0f / sum;
  }
  __syncthreads();

  // --- PV: each wave reduces a 256-key chunk; 4 N-tiles cover dk=0..63 ---
  v8f o_acc[4] = {};
  for (int ks = wave * 256; ks < wave * 256 + 256; ks += 32) {
    v16b ap = load_frag_a(sP + ks, TSEQ, lane);   // DS-path A fragments
#pragma unroll
    for (int j = 0; j < 4; ++j) {
      v16b bvf = load_frag_b(Vb + (size_t)(j * 16) * TSEQ + ks, TSEQ, lane);
      o_acc[j] = wmma_bf16(ap, bvf, o_acc[j]);
    }
  }
#pragma unroll
  for (int j = 0; j < 4; ++j)
#pragma unroll
    for (int v = 0; v < 8; ++v)
      atomicAdd(&sOut[(hh * 8 + v) * 64 + j * 16 + nn], o_acc[j][v]);
  __syncthreads();

  // --- normalize and store [B,T,D] with d = h*64 + dk (bf16 for out-proj) ---
  for (int i = threadIdx.x; i < 16 * 64; i += 256) {
    const int m = i >> 6, d = i & 63;
    attnb[((size_t)b * TSEQ + t0 + m) * DMODEL + h * HDK + d] =
        (bf16)(sOut[i] * sRow[m]);
  }
}

// ---------------------------------------------------------------------------
// Kernel 3: output projection  [4096 x 1024] x Wo^T + bo  -> f32
// ---------------------------------------------------------------------------
__global__ void __launch_bounds__(256)
out_proj(const bf16* __restrict__ attnb, const bf16* __restrict__ wobf,
         const float* __restrict__ bo, float* __restrict__ out) {
  const int lane = threadIdx.x & 31, wave = threadIdx.x >> 5;
  const int mw = wave >> 2, nw = wave & 3;
  const int m0 = blockIdx.x * 64 + mw * 32;
  const int n0 = blockIdx.y * 256 + nw * 64;

  v8f acc[2][4] = {};
  for (int k = 0; k < DMODEL; k += 32) {
    if (k + 32 < DMODEL) {
      __builtin_prefetch((const void*)(attnb + (size_t)m0 * DMODEL + k + 32), 0, 3);
      __builtin_prefetch((const void*)(wobf + (size_t)n0 * DMODEL + k + 32), 0, 3);
    }
    v16b a0 = load_frag_a(attnb + (size_t)m0 * DMODEL + k, DMODEL, lane);
    v16b a1 = load_frag_a(attnb + (size_t)(m0 + 16) * DMODEL + k, DMODEL, lane);
#pragma unroll
    for (int j = 0; j < 4; ++j) {
      v16b bw = load_frag_b(wobf + (size_t)(n0 + j * 16) * DMODEL + k,
                            DMODEL, lane);
      acc[0][j] = wmma_bf16(a0, bw, acc[0][j]);
      acc[1][j] = wmma_bf16(a1, bw, acc[1][j]);
    }
  }

  const int hh = lane >> 4, nn = lane & 15;
#pragma unroll
  for (int mi = 0; mi < 2; ++mi) {
#pragma unroll
    for (int j = 0; j < 4; ++j) {
      const int n = n0 + j * 16 + nn;
      const float bias = bo[n];
#pragma unroll
      for (int v = 0; v < 8; ++v) {
        const int m = m0 + mi * 16 + hh * 8 + v;
        out[(size_t)m * DMODEL + n] = acc[mi][j][v] + bias;
      }
    }
  }
}

// ---------------------------------------------------------------------------
// Host launcher
// ---------------------------------------------------------------------------
extern "C" void kernel_launch(void* const* d_in, const int* in_sizes, int n_in,
                              void* d_out, int out_size, void* d_ws,
                              size_t ws_size, hipStream_t stream) {
  const float* x     = (const float*)d_in[0];
  const float* conf  = (const float*)d_in[1];
  const float* Wq    = (const float*)d_in[2];
  const float* Wk    = (const float*)d_in[3];
  const float* Wv    = (const float*)d_in[4];
  const float* bv    = (const float*)d_in[5];
  const float* Wo    = (const float*)d_in[6];
  const float* bo    = (const float*)d_in[7];
  const float* Wc    = (const float*)d_in[8];
  const float* bc    = (const float*)d_in[9];
  const float* alpha = (const float*)d_in[10];
  float* out = (float*)d_out;

  char* ws = (char*)d_ws;
  bf16* xbf   = (bf16*)(ws);                       // 8 MB
  bf16* wqkv  = (bf16*)(ws + (size_t)(8u << 20));  // 6 MB (packed Q|K|V)
  bf16* wobf  = (bf16*)(ws + (size_t)(14u << 20)); // 2 MB
  bf16* Qd    = (bf16*)(ws + (size_t)(16u << 20)); // 8 MB [B,H,T,DK]
  bf16* Kd    = (bf16*)(ws + (size_t)(24u << 20)); // 8 MB [B,H,T,DK]
  bf16* Vtd   = (bf16*)(ws + (size_t)(32u << 20)); // 8 MB [B,H,DK,T]
  bf16* attnb = (bf16*)(ws + (size_t)(40u << 20)); // 8 MB [B,T,D]

  cvt_kernel<<<(NTOK * DMODEL + 255) / 256, 256, 0, stream>>>(
      x, Wq, Wk, Wv, Wo, xbf, wqkv, wobf);

  proj_qkv<<<dim3(NTOK / 64, 3 * DMODEL / 256), 256, 0, stream>>>(
      xbf, wqkv, bv, Qd, Kd, Vtd);

  attn_kernel<<<BATCH * NHEAD * (TSEQ / 16), 256, ATTN_SMEM, stream>>>(
      Qd, Kd, Vtd, conf, Wc, bc, alpha, attnb);

  out_proj<<<dim3(NTOK / 64, DMODEL / 256), 256, 0, stream>>>(
      attnb, wobf, bo, out);
}
